// Head_62646392979495
// MI455X (gfx1250) — matrix operand order repaired and checked by reading, not compiled, not run on hardware
//
#include <hip/hip_runtime.h>
#include <hip/hip_bf16.h>

typedef __bf16 bf16_t;
typedef __attribute__((ext_vector_type(16))) __bf16 v16bf;
typedef __attribute__((ext_vector_type(8)))  __bf16 v8bf;
typedef __attribute__((ext_vector_type(8)))  float  v8f;

union V16 { v16bf v; v8bf h[2]; };

#define BATCH 8
#define TLEN  2048
#define EMB   1024
#define HEAD  128
#define SCALE 0.08838834764831845f  // 1/sqrt(128)

static __device__ inline v8f wmma_bf16(const V16& a, const V16& b, v8f c) {
    return __builtin_amdgcn_wmma_f32_16x16x32_bf16(
        /*neg_a=*/false, a.v, /*neg_b=*/false, b.v,
        /*c_mod=*/(short)0, c, /*reuse_a=*/false, /*reuse_b=*/false);
}

// ---------------------------------------------------------------------------
// Kernel 0: convert x f32 -> bf16 once (8 elems/thread, b128 in / b128 out).
// xb [B*T, E] bf16 (32 MB -> resident in 192 MB L2 for all proj re-reads).
// ---------------------------------------------------------------------------
__global__ __launch_bounds__(256) void xprep_kernel(const float* __restrict__ x,
                                                    bf16_t* __restrict__ xb) {
    const size_t base = ((size_t)blockIdx.x * blockDim.x + threadIdx.x) * 8;
    float4 a = *(const float4*)(x + base);
    float4 b = *(const float4*)(x + base + 4);
    v8bf o;
    o[0] = (bf16_t)a.x; o[1] = (bf16_t)a.y; o[2] = (bf16_t)a.z; o[3] = (bf16_t)a.w;
    o[4] = (bf16_t)b.x; o[5] = (bf16_t)b.y; o[6] = (bf16_t)b.z; o[7] = (bf16_t)b.w;
    *(v8bf*)(xb + base) = o;
}

// ---------------------------------------------------------------------------
// Kernel 1: transpose + convert W (f32 [E,H]) -> WT (bf16 [3][H][E])
// order: 0=Wk, 1=Wq, 2=Wv
// ---------------------------------------------------------------------------
__global__ __launch_bounds__(256) void wprep_kernel(const float* __restrict__ Wk,
                                                    const float* __restrict__ Wq,
                                                    const float* __restrict__ Wv,
                                                    bf16_t* __restrict__ WT) {
    int idx = blockIdx.x * blockDim.x + threadIdx.x;
    if (idx >= 3 * HEAD * EMB) return;
    int m = idx / (HEAD * EMB);
    int r = idx % (HEAD * EMB);
    int h = r / EMB;
    int e = r % EMB;
    const float* W = (m == 0) ? Wk : ((m == 1) ? Wq : Wv);
    WT[idx] = (bf16_t)W[e * HEAD + h];
}

// ---------------------------------------------------------------------------
// Kernel 2: projections. Each wave computes a 16x128 output row-block
// (8 accumulators) of xb@W via WMMA over K=1024 in chunks of 32; one A-tile
// load feeds 8 WMMAs.
//   m==0 -> Kb  bf16 [B*T, H]
//   m==1 -> Qb  bf16 [B*T, H]  (pre-scaled by 1/sqrt(H))
//   m==2 -> VTb bf16 [B][H][T] (transposed store)
// ---------------------------------------------------------------------------
__global__ __launch_bounds__(128) void proj_kernel(const bf16_t* __restrict__ xb,
                                                   const bf16_t* __restrict__ WT,
                                                   bf16_t* __restrict__ Qb,
                                                   bf16_t* __restrict__ Kb,
                                                   bf16_t* __restrict__ VTb) {
    const int lane = threadIdx.x & 31;
    const int wave = threadIdx.x >> 5;
    const int tile = blockIdx.x * 4 + wave;          // 0 .. 3071
    const int m  = tile / (BATCH * TLEN / 16);       // which weight matrix
    const int tt = tile % (BATCH * TLEN / 16);       // 16-row tile over B*T

    const int col  = lane & 15;
    const int half = lane >> 4;

    const bf16_t* xrow  = xb + (size_t)(tt * 16 + col) * EMB;     // A row
    const bf16_t* wbase = WT + (size_t)m * HEAD * EMB;

    v8f acc[8];
    #pragma unroll
    for (int j = 0; j < 8; ++j) acc[j] = (v8f){};

    for (int c = 0; c < EMB / 32; ++c) {
        const int ec = c * 32;
        V16 a;
        a.h[0] = *(const v8bf*)(xrow + ec + 8 * half);
        a.h[1] = *(const v8bf*)(xrow + ec + 16 + 8 * half);
        #pragma unroll
        for (int j = 0; j < 8; ++j) {
            const bf16_t* wc = wbase + (size_t)(j * 16 + col) * EMB + ec;
            V16 bm;
            bm.h[0] = *(const v8bf*)(wc + 8 * half);
            bm.h[1] = *(const v8bf*)(wc + 16 + 8 * half);
            acc[j] = wmma_bf16(a, bm, acc[j]);
        }
    }

    if (m == 2) {
        // V transposed: VTb[b][h][t]; lane column = channel, rows contiguous t
        const int t0   = tt * 16;
        const int b    = t0 / TLEN;
        const int tloc = t0 - b * TLEN;
        #pragma unroll
        for (int j = 0; j < 8; ++j) {
            v8bf pk;
            #pragma unroll
            for (int v = 0; v < 8; ++v) pk[v] = (bf16_t)acc[j][v];
            *(v8bf*)(VTb + (size_t)b * HEAD * TLEN
                         + (size_t)(j * 16 + col) * TLEN
                         + tloc + 8 * half) = pk;
        }
    } else {
        bf16_t* dst = (m == 1) ? Qb : Kb;
        const float s = (m == 1) ? SCALE : 1.0f;
        #pragma unroll
        for (int j = 0; j < 8; ++j)
            #pragma unroll
            for (int v = 0; v < 8; ++v) {
                const int row = tt * 16 + v + 8 * half;  // C layout: M = v+8*half
                dst[(size_t)row * HEAD + j * 16 + col] = (bf16_t)(acc[j][v] * s);
            }
    }
}

// ---------------------------------------------------------------------------
// Kernel 3: attention. One wave per 16-query tile; flash-style loop over
// 64-key tiles with online softmax; P converted C-layout -> A-layout via a
// per-wave LDS round trip ordered by s_wait_dscnt (same-wave DS is in-order).
// ---------------------------------------------------------------------------
__global__ __launch_bounds__(128) void attn_kernel(const bf16_t* __restrict__ Qb,
                                                   const bf16_t* __restrict__ Kb,
                                                   const bf16_t* __restrict__ VTb,
                                                   float* __restrict__ out) {
    __shared__ __align__(16) bf16_t plds[4 * 16 * 64];

    const int lane = threadIdx.x & 31;
    const int wave = threadIdx.x >> 5;
    const int idx  = blockIdx.x * 4 + wave;    // 0 .. 1023
    const int b    = idx >> 7;                 // / (T/16)
    const int qt   = idx & 127;
    const int q0   = qt * 16;
    const int col  = lane & 15;
    const int half = lane >> 4;

    // Q A-tiles (held in registers for the whole kernel), already scaled.
    const bf16_t* qrow = Qb + ((size_t)b * TLEN + q0 + col) * HEAD;
    V16 aq[4];
    #pragma unroll
    for (int c = 0; c < 4; ++c) {
        aq[c].h[0] = *(const v8bf*)(qrow + 32 * c + 8 * half);
        aq[c].h[1] = *(const v8bf*)(qrow + 32 * c + 16 + 8 * half);
    }

    v8f o[8];
    #pragma unroll
    for (int j = 0; j < 8; ++j) o[j] = (v8f){};
    float mrun[8], lrun[8];
    #pragma unroll
    for (int v = 0; v < 8; ++v) { mrun[v] = -1e30f; lrun[v] = 0.0f; }

    const bf16_t* kbase = Kb  + (size_t)b * TLEN * HEAD;
    const bf16_t* vbase = VTb + (size_t)b * HEAD * TLEN;
    bf16_t* pl = plds + wave * 16 * 64;

    for (int kt = 0; kt < TLEN / 64; ++kt) {
        const int k0 = kt * 64;

        // prefetch next K tile (global_prefetch_b8) behind this tile's VALU work
        if (kt + 1 < TLEN / 64)
            __builtin_prefetch(kbase + (size_t)(k0 + 64 + col) * HEAD, 0, 3);

        // ---- S = Q @ K^T : four 16x16 C tiles (keys k0+16t .. k0+16t+15) ----
        v8f s[4];
        #pragma unroll
        for (int t = 0; t < 4; ++t) s[t] = (v8f){};
        #pragma unroll
        for (int c = 0; c < 4; ++c)
            #pragma unroll
            for (int t = 0; t < 4; ++t) {
                const bf16_t* kp = kbase + (size_t)(k0 + 16 * t + col) * HEAD
                                         + 32 * c;
                V16 bk;
                bk.h[0] = *(const v8bf*)(kp + 8 * half);
                bk.h[1] = *(const v8bf*)(kp + 16 + 8 * half);
                s[t] = wmma_bf16(aq[c], bk, s[t]);
            }

        // ---- online softmax (rows = v + 8*half, cols across 16 lanes) ----
        float alpha[8], p[4][8];
        #pragma unroll
        for (int v = 0; v < 8; ++v) {
            float t = fmaxf(fmaxf(s[0][v], s[1][v]), fmaxf(s[2][v], s[3][v]));
            t = fmaxf(t, __shfl_xor(t, 1, 32));
            t = fmaxf(t, __shfl_xor(t, 2, 32));
            t = fmaxf(t, __shfl_xor(t, 4, 32));
            t = fmaxf(t, __shfl_xor(t, 8, 32));
            const float mn = fmaxf(mrun[v], t);
            alpha[v] = __expf(mrun[v] - mn);
            mrun[v]  = mn;
            #pragma unroll
            for (int g = 0; g < 4; ++g) p[g][v] = __expf(s[g][v] - mn);
            float r = (p[0][v] + p[1][v]) + (p[2][v] + p[3][v]);
            r += __shfl_xor(r, 1, 32);
            r += __shfl_xor(r, 2, 32);
            r += __shfl_xor(r, 4, 32);
            r += __shfl_xor(r, 8, 32);
            lrun[v] = lrun[v] * alpha[v] + r;
        }
        #pragma unroll
        for (int j = 0; j < 8; ++j)
            #pragma unroll
            for (int v = 0; v < 8; ++v) o[j][v] *= alpha[v];

        // ---- P: C layout -> A layout via per-wave LDS transpose (16x64) ----
        #pragma unroll
        for (int g = 0; g < 4; ++g)
            #pragma unroll
            for (int v = 0; v < 8; ++v) {
                const int rq = v + 8 * half;
                pl[rq * 64 + g * 16 + col] = (bf16_t)p[g][v];
            }
        asm volatile("s_wait_dscnt 0" ::: "memory");  // same-wave DS in-order
        V16 ap[2];
        #pragma unroll
        for (int g = 0; g < 2; ++g) {
            ap[g].h[0] = *(const v8bf*)(pl + col * 64 + 32 * g + 8 * half);
            ap[g].h[1] = *(const v8bf*)(pl + col * 64 + 32 * g + 16 + 8 * half);
        }

        // ---- O += P @ V (V^T layout -> contiguous B-tile loads) ----
        #pragma unroll
        for (int j = 0; j < 8; ++j) {
            const bf16_t* vp = vbase + (size_t)(j * 16 + col) * TLEN + k0;
            #pragma unroll
            for (int g = 0; g < 2; ++g) {
                V16 bv;
                bv.h[0] = *(const v8bf*)(vp + 32 * g + 8 * half);
                bv.h[1] = *(const v8bf*)(vp + 32 * g + 16 + 8 * half);
                o[j] = wmma_bf16(ap[g], bv, o[j]);
            }
        }
        asm volatile("" ::: "memory");  // keep LDS loads before next iter's stores
    }

    // ---- epilogue: normalize and store f32 ----
    #pragma unroll
    for (int v = 0; v < 8; ++v) {
        const float inv = 1.0f / lrun[v];
        const int rq = v + 8 * half;
        float* orow = out + ((size_t)b * TLEN + q0 + rq) * HEAD + col;
        #pragma unroll
        for (int j = 0; j < 8; ++j) orow[j * 16] = o[j][v] * inv;
    }
}

// ---------------------------------------------------------------------------
extern "C" void kernel_launch(void* const* d_in, const int* in_sizes, int n_in,
                              void* d_out, int out_size, void* d_ws, size_t ws_size,
                              hipStream_t stream) {
    const float* x  = (const float*)d_in[0];
    const float* Wk = (const float*)d_in[1];
    const float* Wq = (const float*)d_in[2];
    const float* Wv = (const float*)d_in[3];
    float* out = (float*)d_out;

    // workspace layout (bf16):
    //   WT  : 3*H*E   (0.75 MB)
    //   xb  : B*T*E   (32 MB)
    //   Qb  : B*T*H   (4 MB)
    //   Kb  : B*T*H   (4 MB)
    //   VTb : B*H*T   (4 MB)
    char* w = (char*)d_ws;
    bf16_t* WT  = (bf16_t*)w;
    bf16_t* xb  = WT + (size_t)3 * HEAD * EMB;
    bf16_t* Qb  = xb + (size_t)BATCH * TLEN * EMB;
    bf16_t* Kb  = Qb + (size_t)BATCH * TLEN * HEAD;
    bf16_t* VTb = Kb + (size_t)BATCH * TLEN * HEAD;

    // 0) x f32 -> bf16 (done once; proj re-reads hit L2)
    {
        const size_t n = (size_t)BATCH * TLEN * EMB / 8;   // threads
        xprep_kernel<<<(int)(n / 256), 256, 0, stream>>>(x, xb);
    }
    // 1) W transpose+convert
    {
        const int n = 3 * HEAD * EMB;
        wprep_kernel<<<(n + 255) / 256, 256, 0, stream>>>(Wk, Wq, Wv, WT);
    }
    // 2) projections: 3 * (B*T/16) = 3072 row-block waves, 4 waves/block
    proj_kernel<<<3072 / 4, 128, 0, stream>>>(xb, WT, Qb, Kb, VTb);
    // 3) attention: B * T/16 = 1024 query tiles, 4 waves/block
    attn_kernel<<<1024 / 4, 128, 0, stream>>>(Qb, Kb, VTb, out);
}
